// DiagRNN_33011118637315
// MI455X (gfx1250) — matrix operand order repaired and checked by reading, not compiled, not run on hardware
//
#include <hip/hip_runtime.h>

#define L_SEQ  16384
#define DM     1024
#define DH     1024
#define CHUNKS 128
#define TCH    (L_SEQ / CHUNKS)   // 128 steps per chunk

typedef __attribute__((ext_vector_type(16))) __bf16 v16bf;
typedef __attribute__((ext_vector_type(8)))  float  v8f;

struct __attribute__((aligned(16))) U128 { unsigned int a, b, c, d; };
union FragBF { v16bf v; U128 q[2]; };
union AccU   { v8f v; float f[8]; };

__device__ __forceinline__ unsigned short f2bf(float x) {
  union { float f; unsigned u; } v; v.f = x;
  unsigned r = v.u + 0x7FFFu + ((v.u >> 16) & 1u);   // round-to-nearest-even
  return (unsigned short)(r >> 16);
}
__device__ __forceinline__ float bf2f(unsigned short b) {
  union { unsigned u; float f; } v; v.u = ((unsigned)b) << 16; return v.f;
}

// ---------------- fp32 -> (hi, lo) bf16 split ----------------
__global__ void split_bf16x2(const float* __restrict__ x,
                             unsigned short* __restrict__ hi,
                             unsigned short* __restrict__ lo, int n) {
  int i = blockIdx.x * blockDim.x + threadIdx.x;
  int stride = gridDim.x * blockDim.x;
  for (; i < n; i += stride) {
    float v = x[i];
    unsigned short h = f2bf(v);
    hi[i] = h;
    lo[i] = f2bf(v - bf2f(h));
  }
}

// ---------------- compensated-bf16 WMMA GEMM: D = A * B^T (+ Dvec*Xres) ----
// A: M x K (row-major bf16 hi/lo planes), B: N x K (row-major bf16 hi/lo)
// Block = 256 threads = 8 waves; block tile 128x128; wave tile 32x64.
__global__ void __launch_bounds__(256)
gemm_bf16x3(const unsigned short* __restrict__ Ahi,
            const unsigned short* __restrict__ Alo,
            const unsigned short* __restrict__ Bhi,
            const unsigned short* __restrict__ Blo,
            float* __restrict__ Dst,
            int M, int N, int K,
            const float* __restrict__ Xres,   // optional epilogue: += Dvec[n]*Xres[m,n]
            const float* __restrict__ Dvec) {
  const int tid  = threadIdx.x;
  const int lane = tid & 31;
  const int wave = tid >> 5;
  const int half = lane >> 4;     // 0/1: which half-wave
  const int l16  = lane & 15;

  const int wm   = wave >> 1;                        // 0..3 along M
  const int wn   = wave & 1;                         // 0..1 along N
  const int row0 = blockIdx.x * 128 + wm * 32;
  const int col0 = blockIdx.y * 128 + wn * 64;

  const v8f vzero = {0.f,0.f,0.f,0.f,0.f,0.f,0.f,0.f};
  v8f acc[2][4];
#pragma unroll
  for (int i = 0; i < 2; ++i)
#pragma unroll
    for (int j = 0; j < 4; ++j) acc[i][j] = vzero;

  const size_t ldk = (size_t)K;

  for (int kk = 0; kk < K; kk += 32) {
    FragBF ah[2], al[2], bh[4], bl[4];
    // A 16x32 bf16 fragment (ISA §7.12.2): lane holds row (row0+16i+l16),
    // K chunks [kk+8*half, +8) and [kk+16+8*half, +8) -> two b128 loads.
#pragma unroll
    for (int i = 0; i < 2; ++i) {
      const unsigned short* p = Ahi + (size_t)(row0 + 16*i + l16) * ldk + kk + 8*half;
      const unsigned short* q = Alo + (size_t)(row0 + 16*i + l16) * ldk + kk + 8*half;
      ah[i].q[0] = *(const U128*)p;   ah[i].q[1] = *(const U128*)(p + 16);
      al[i].q[0] = *(const U128*)q;   al[i].q[1] = *(const U128*)(q + 16);
    }
    // B 32x16 bf16 fragment: lane holds col (col0+16j+l16),
    // K chunks [kk+8*half, +8) and [kk+16+8*half, +8) -> two b128 loads (B^T is row n of B).
#pragma unroll
    for (int j = 0; j < 4; ++j) {
      const unsigned short* p = Bhi + (size_t)(col0 + 16*j + l16) * ldk + kk + 8*half;
      const unsigned short* q = Blo + (size_t)(col0 + 16*j + l16) * ldk + kk + 8*half;
      bh[j].q[0] = *(const U128*)p;   bh[j].q[1] = *(const U128*)(p + 16);
      bl[j].q[0] = *(const U128*)q;   bl[j].q[1] = *(const U128*)(q + 16);
    }
#pragma unroll
    for (int i = 0; i < 2; ++i)
#pragma unroll
      for (int j = 0; j < 4; ++j) {
        acc[i][j] = __builtin_amdgcn_wmma_f32_16x16x32_bf16(
            false, ah[i].v, false, bh[j].v, (short)0, acc[i][j], false, false);
        acc[i][j] = __builtin_amdgcn_wmma_f32_16x16x32_bf16(
            false, ah[i].v, false, bl[j].v, (short)0, acc[i][j], false, false);
        acc[i][j] = __builtin_amdgcn_wmma_f32_16x16x32_bf16(
            false, al[i].v, false, bh[j].v, (short)0, acc[i][j], false, false);
      }
  }

  // C/D layout: col = l16; row-in-tile = 8*half + r (VGPR r).
#pragma unroll
  for (int i = 0; i < 2; ++i)
#pragma unroll
    for (int j = 0; j < 4; ++j) {
      AccU a; a.v = acc[i][j];
      const int n = col0 + 16*j + l16;
#pragma unroll
      for (int r = 0; r < 8; ++r) {
        const int m = row0 + 16*i + 8*half + r;
        float val = a.f[r];
        if (Xres) val += Dvec[n] * Xres[(size_t)m * N + n];
        Dst[(size_t)m * N + n] = val;
      }
    }
}

// ---------------- chunked complex diagonal scan ----------------
// Phase 1: local scan of each chunk from zero; store end state S[c,h].
__global__ void scan_phase1(const float* __restrict__ BuRe, const float* __restrict__ BuIm,
                            const float* __restrict__ Are,  const float* __restrict__ Aim,
                            float* __restrict__ Sre, float* __restrict__ Sim) {
  const int h = blockIdx.x * blockDim.x + threadIdx.x;
  const int c = blockIdx.y;
  const float ar = Are[h], ai = Aim[h];
  float sr = 0.f, si = 0.f;
  size_t base = (size_t)c * TCH * DH + h;
  for (int t = 0; t < TCH; ++t) {
    const float br = BuRe[base], bi = BuIm[base];
    const float nr = ar * sr - ai * si + br;
    const float ni = ar * si + ai * sr + bi;
    sr = nr; si = ni;
    base += DH;
  }
  Sre[(size_t)c * DH + h] = sr;
  Sim[(size_t)c * DH + h] = si;
}

// Phase 2: per-channel sequential combine across CHUNKS (exclusive prefix P[c,h]).
__global__ void scan_phase2(const float* __restrict__ Are, const float* __restrict__ Aim,
                            const float* __restrict__ Sre, const float* __restrict__ Sim,
                            float* __restrict__ Pre, float* __restrict__ Pim) {
  const int h = threadIdx.x;                 // blockDim = 1024
  const float ar = Are[h], ai = Aim[h];
  float pr = 1.f, pi = 0.f;                  // A^TCH
  for (int t = 0; t < TCH; ++t) {
    const float nr = pr * ar - pi * ai;
    const float ni = pr * ai + pi * ar;
    pr = nr; pi = ni;
  }
  float xr = 0.f, xi = 0.f;
  for (int c = 0; c < CHUNKS; ++c) {
    Pre[(size_t)c * DH + h] = xr;
    Pim[(size_t)c * DH + h] = xi;
    const float sr = Sre[(size_t)c * DH + h], si = Sim[(size_t)c * DH + h];
    const float nr = pr * xr - pi * xi + sr;
    const float ni = pr * xi + pi * xr + si;
    xr = nr; xi = ni;
  }
}

// Phase 3: seeded replay; emit Re(h) as (hi,lo) bf16 planes for the WMMA epilogue GEMM.
__global__ void scan_phase3(const float* __restrict__ BuRe, const float* __restrict__ BuIm,
                            const float* __restrict__ Are,  const float* __restrict__ Aim,
                            const float* __restrict__ Pre,  const float* __restrict__ Pim,
                            unsigned short* __restrict__ Hhi, unsigned short* __restrict__ Hlo) {
  const int h = blockIdx.x * blockDim.x + threadIdx.x;
  const int c = blockIdx.y;
  const float ar = Are[h], ai = Aim[h];
  float sr = Pre[(size_t)c * DH + h], si = Pim[(size_t)c * DH + h];
  size_t base = (size_t)c * TCH * DH + h;
  for (int t = 0; t < TCH; ++t) {
    const float br = BuRe[base], bi = BuIm[base];
    const float nr = ar * sr - ai * si + br;
    const float ni = ar * si + ai * sr + bi;
    sr = nr; si = ni;
    const unsigned short hb = f2bf(sr);
    Hhi[base] = hb;
    Hlo[base] = f2bf(sr - bf2f(hb));
    base += DH;
  }
}

extern "C" void kernel_launch(void* const* d_in, const int* in_sizes, int n_in,
                              void* d_out, int out_size, void* d_ws, size_t ws_size,
                              hipStream_t stream) {
  (void)in_sizes; (void)n_in; (void)out_size; (void)ws_size;
  const float* X   = (const float*)d_in[0];
  const float* Are = (const float*)d_in[1];
  const float* Aim = (const float*)d_in[2];
  const float* Bre = (const float*)d_in[3];
  const float* Bim = (const float*)d_in[4];
  const float* Cm  = (const float*)d_in[5];
  const float* Dv  = (const float*)d_in[6];
  float* Y = (float*)d_out;

  char* ws = (char*)d_ws;
  size_t off = 0;
  auto carve = [&](size_t bytes) -> void* {
    void* p = ws + off;
    off = (off + bytes + 255) & ~(size_t)255;
    return p;
  };

  unsigned short* Xhi   = (unsigned short*)carve((size_t)L_SEQ * DM * 2);
  unsigned short* Xlo   = (unsigned short*)carve((size_t)L_SEQ * DM * 2);
  unsigned short* BreHi = (unsigned short*)carve((size_t)DH * DM * 2);
  unsigned short* BreLo = (unsigned short*)carve((size_t)DH * DM * 2);
  unsigned short* BimHi = (unsigned short*)carve((size_t)DH * DM * 2);
  unsigned short* BimLo = (unsigned short*)carve((size_t)DH * DM * 2);
  unsigned short* Chi   = (unsigned short*)carve((size_t)DM * DH * 2);
  unsigned short* Clo   = (unsigned short*)carve((size_t)DM * DH * 2);
  float* BuRe = (float*)carve((size_t)L_SEQ * DH * 4);
  float* BuIm = (float*)carve((size_t)L_SEQ * DH * 4);
  unsigned short* Hhi = (unsigned short*)carve((size_t)L_SEQ * DH * 2);
  unsigned short* Hlo = (unsigned short*)carve((size_t)L_SEQ * DH * 2);
  float* Sre = (float*)carve((size_t)CHUNKS * DH * 4);
  float* Sim = (float*)carve((size_t)CHUNKS * DH * 4);
  float* Pre = (float*)carve((size_t)CHUNKS * DH * 4);
  float* Pim = (float*)carve((size_t)CHUNKS * DH * 4);

  // 1) split fp32 operands into compensated bf16 planes
  split_bf16x2<<<dim3(1024), 256, 0, stream>>>(X,   Xhi,   Xlo,   L_SEQ * DM);
  split_bf16x2<<<dim3(512),  256, 0, stream>>>(Bre, BreHi, BreLo, DH * DM);
  split_bf16x2<<<dim3(512),  256, 0, stream>>>(Bim, BimHi, BimLo, DH * DM);
  split_bf16x2<<<dim3(512),  256, 0, stream>>>(Cm,  Chi,   Clo,   DM * DH);

  // 2) Bu_re = X @ Bre^T, Bu_im = X @ Bim^T  (WMMA bf16x3)
  dim3 gB(L_SEQ / 128, DH / 128);
  gemm_bf16x3<<<gB, 256, 0, stream>>>(Xhi, Xlo, BreHi, BreLo, BuRe,
                                      L_SEQ, DH, DM, nullptr, nullptr);
  gemm_bf16x3<<<gB, 256, 0, stream>>>(Xhi, Xlo, BimHi, BimLo, BuIm,
                                      L_SEQ, DH, DM, nullptr, nullptr);

  // 3) chunked complex diagonal scan
  scan_phase1<<<dim3(DH / 256, CHUNKS), 256, 0, stream>>>(BuRe, BuIm, Are, Aim, Sre, Sim);
  scan_phase2<<<1, 1024, 0, stream>>>(Are, Aim, Sre, Sim, Pre, Pim);
  scan_phase3<<<dim3(DH / 256, CHUNKS), 256, 0, stream>>>(BuRe, BuIm, Are, Aim,
                                                          Pre, Pim, Hhi, Hlo);

  // 4) Y = Re(h) @ C^T + D * X  (WMMA bf16x3 with fused elementwise epilogue)
  dim3 gY(L_SEQ / 128, DM / 128);
  gemm_bf16x3<<<gY, 256, 0, stream>>>(Hhi, Hlo, Chi, Clo, Y,
                                      L_SEQ, DM, DH, X, Dv);
}
